// GAT_74904229642412
// MI455X (gfx1250) — compile-verified, matrix operand order checked by reading, not run on hardware
//
#include <hip/hip_runtime.h>
#include <hip/hip_bf16.h>
#include <math.h>

#define NNODES   8192
#define INCH     256
#define NHEADS   8
#define CPH      32
#define HC       256            // NHEADS*CPH
#define NEDGE    262144
#define EPRIME   (NEDGE + NNODES)
#define OUTW_COLS (NNODES * HC) // 2097152
#define NEG_SLOPE 0.2f

typedef __attribute__((ext_vector_type(16))) __bf16 v16bf;
typedef __attribute__((ext_vector_type(8)))  float  v8f;

// ---------- helpers ----------
__device__ __forceinline__ unsigned fenc(float f) {
    unsigned u = __float_as_uint(f);
    return (u & 0x80000000u) ? ~u : (u | 0x80000000u);
}
__device__ __forceinline__ float fdec(unsigned u) {
    unsigned v = (u & 0x80000000u) ? (u & 0x7FFFFFFFu) : ~u;
    return __uint_as_float(v);
}
__device__ __forceinline__ void edge_sd(const int* __restrict__ ei, int e, int& s, int& d) {
    if (e < NEDGE) { s = ei[e]; d = ei[NEDGE + e]; }
    else           { s = e - NEDGE; d = s; }          // self loops appended
}
__device__ __forceinline__ float lrelu(float v) { return v > 0.f ? v : NEG_SLOPE * v; }

// ---------- 1. projection GEMM: xh = x @ lin_w^T  (bf16 WMMA, f32 accum) ----------
// grid (512, 4), block 32: one wave -> 16x64 output strip.
// All 8 K-fragments of A stay register-resident (64 VGPRs) and are reused
// across 4 B tiles -> 32 WMMAs per wave, x traffic cut 4x vs tile-per-wave.
__global__ __launch_bounds__(32)
void gat_proj_wmma(const float* __restrict__ x, const float* __restrict__ w,
                   float* __restrict__ xh) {
    const int lane  = threadIdx.x & 31;
    const int m0    = blockIdx.x * 16;
    const int n0    = blockIdx.y * 64;
    const int row   = m0 + (lane & 15);          // A row per lane
    const int khalf = (lane >> 4) * 8;           // lane-half K offset per ISA layout

    // Load the full 16x256 A slab once as 8 bf16 fragments.
    v16bf afrag[8];
#pragma unroll
    for (int t = 0; t < 8; ++t) {
#pragma unroll
        for (int p = 0; p < 8; ++p) {
            const int k = t * 32 + ((p < 4) ? 0 : 16) + khalf + (p & 3) * 2;
            const float2 xa = *reinterpret_cast<const float2*>(x + row * INCH + k);
            afrag[t][2 * p]     = (__bf16)xa.x;
            afrag[t][2 * p + 1] = (__bf16)xa.y;
        }
    }

#pragma unroll
    for (int nt = 0; nt < 4; ++nt) {
        const int col = n0 + nt * 16 + (lane & 15);   // B column per lane (= lin_w row)
        v8f acc = {};
#pragma unroll
        for (int t = 0; t < 8; ++t) {
            v16bf b;
#pragma unroll
            for (int p = 0; p < 8; ++p) {
                const int k = t * 32 + ((p < 4) ? 0 : 16) + khalf + (p & 3) * 2;
                const float2 wb = *reinterpret_cast<const float2*>(w + col * INCH + k);
                b[2 * p]     = (__bf16)wb.x;
                b[2 * p + 1] = (__bf16)wb.y;
            }
            acc = __builtin_amdgcn_wmma_f32_16x16x32_bf16(false, afrag[t], false, b,
                                                          (short)0, acc, false, false);
        }
        const int rbase = m0 + ((lane >> 4) << 3);    // C/D layout: M = r + 8*(lane>=16)
#pragma unroll
        for (int r = 0; r < 8; ++r)
            xh[(rbase + r) * HC + col] = acc[r];
    }
}

// ---------- 2. per-node attention scalars a_src/a_dst ----------
__global__ void gat_att(const float* __restrict__ xh,
                        const float* __restrict__ att_src, const float* __restrict__ att_dst,
                        float* __restrict__ a_src, float* __restrict__ a_dst) {
    const int idx = blockIdx.x * blockDim.x + threadIdx.x;  // n*H + h
    if (idx >= NNODES * NHEADS) return;
    const int h = idx & (NHEADS - 1);
    const float* v  = xh + (idx >> 3) * HC + h * CPH;
    const float* ws = att_src + h * CPH;
    const float* wd = att_dst + h * CPH;
    float s = 0.f, d = 0.f;
#pragma unroll
    for (int c = 0; c < CPH; ++c) { const float t = v[c]; s += t * ws[c]; d += t * wd[c]; }
    a_src[idx] = s; a_dst[idx] = d;
}

// ---------- 3a. segment max over incoming edges (encoded u32 atomic max) ----------
__global__ void gat_edge_max(const int* __restrict__ ei,
                             const float* __restrict__ a_src, const float* __restrict__ a_dst,
                             unsigned* __restrict__ m_enc) {
    const int e = blockIdx.x * blockDim.x + threadIdx.x;
    if (e >= EPRIME) return;
    int s, d; edge_sd(ei, e, s, d);
#pragma unroll
    for (int h = 0; h < NHEADS; ++h) {
        const float v = lrelu(a_src[s * NHEADS + h] + a_dst[d * NHEADS + h]);
        atomicMax(&m_enc[d * NHEADS + h], fenc(v));
    }
}

// ---------- 3b. segment sum of exp(e - max) ----------
__global__ void gat_edge_expsum(const int* __restrict__ ei,
                                const float* __restrict__ a_src, const float* __restrict__ a_dst,
                                const unsigned* __restrict__ m_enc, float* __restrict__ denom) {
    const int e = blockIdx.x * blockDim.x + threadIdx.x;
    if (e >= EPRIME) return;
    int s, d; edge_sd(ei, e, s, d);
#pragma unroll
    for (int h = 0; h < NHEADS; ++h) {
        const float v  = lrelu(a_src[s * NHEADS + h] + a_dst[d * NHEADS + h]);
        const float ex = __expf(v - fdec(m_enc[d * NHEADS + h]));
        atomicAdd(&denom[d * NHEADS + h], ex);
    }
}

// ---------- 4. weighted scatter-add aggregation: agg[dst] += alpha * xh[src] ----------
// grid EPRIME, block 256: one block per edge, thread q -> (h = q>>5, c = q&31)
__global__ __launch_bounds__(256)
void gat_aggregate(const int* __restrict__ ei,
                   const float* __restrict__ a_src, const float* __restrict__ a_dst,
                   const unsigned* __restrict__ m_enc, const float* __restrict__ denom,
                   const float* __restrict__ xh, float* __restrict__ agg) {
    const int e = blockIdx.x;
    const int q = threadIdx.x;
    int s, d; edge_sd(ei, e, s, d);
    const int h = q >> 5;
    const float v     = lrelu(a_src[s * NHEADS + h] + a_dst[d * NHEADS + h]);
    const float alpha = __expf(v - fdec(m_enc[d * NHEADS + h])) / denom[d * NHEADS + h];
    atomicAdd(&agg[d * HC + q], alpha * xh[s * HC + q]);
}

// ---------- 5. output layer: y[c] = sum_i (agg[i]+bias) * out_w[c][i] ----------
// Dominant roofline term: streams 268 MB of out_w once (non-temporal, ~11.5us
// at 23.3 TB/s), 32 register accumulators, wave32 shuffle reduction.
__global__ __launch_bounds__(256)
void gat_outdot(const float* __restrict__ agg, const float* __restrict__ bias,
                const float* __restrict__ out_w, float* __restrict__ ypart) {
    float acc[CPH];
#pragma unroll
    for (int c = 0; c < CPH; ++c) acc[c] = 0.f;
    const int stride = gridDim.x * blockDim.x;
    for (int i = blockIdx.x * blockDim.x + threadIdx.x; i < OUTW_COLS; i += stride) {
        const float o = agg[i] + bias[i & (HC - 1)];
#pragma unroll
        for (int c = 0; c < CPH; ++c)
            acc[c] += o * __builtin_nontemporal_load(out_w + (size_t)c * OUTW_COLS + i);
    }
#pragma unroll
    for (int c = 0; c < CPH; ++c) {
        float r = acc[c];
#pragma unroll
        for (int off = 16; off > 0; off >>= 1) r += __shfl_xor(r, off, 32);  // wave32
        if ((threadIdx.x & 31) == 0) atomicAdd(&ypart[c], r);
    }
}

// ---------- 6. 32-wide softmax on one wave ----------
__global__ void gat_softmax(const float* __restrict__ ypart, const float* __restrict__ out_b,
                            float* __restrict__ out) {
    const int c = threadIdx.x;                   // 32 threads
    const float v = ypart[c] + out_b[c];
    float mx = v;
#pragma unroll
    for (int off = 16; off > 0; off >>= 1) mx = fmaxf(mx, __shfl_xor(mx, off, 32));
    const float e = __expf(v - mx);
    float sm = e;
#pragma unroll
    for (int off = 16; off > 0; off >>= 1) sm += __shfl_xor(sm, off, 32);
    out[c] = e / sm;
}

// ---------- host ----------
extern "C" void kernel_launch(void* const* d_in, const int* in_sizes, int n_in,
                              void* d_out, int out_size, void* d_ws, size_t ws_size,
                              hipStream_t stream) {
    const float* x       = (const float*)d_in[0];
    const int*   ei      = (const int*)  d_in[1];
    const float* lin_w   = (const float*)d_in[2];
    const float* att_src = (const float*)d_in[3];
    const float* att_dst = (const float*)d_in[4];
    const float* bias    = (const float*)d_in[5];
    const float* out_w   = (const float*)d_in[6];
    const float* out_b   = (const float*)d_in[7];
    float* out = (float*)d_out;

    // workspace layout (zero-init region is contiguous: m_enc..ypart)
    float*    xh     = (float*)d_ws;                         // N*HC
    float*    a_src  = xh + (size_t)NNODES * HC;             // N*H
    float*    a_dst  = a_src + NNODES * NHEADS;              // N*H
    unsigned* m_enc  = (unsigned*)(a_dst + NNODES * NHEADS); // N*H (0 == encoded max identity)
    float*    denom  = (float*)(m_enc + NNODES * NHEADS);    // N*H
    float*    agg    = denom + NNODES * NHEADS;              // N*HC
    float*    ypart  = agg + (size_t)NNODES * HC;            // 32

    const size_t zero_bytes =
        (size_t)(2 * NNODES * NHEADS + NNODES * HC + CPH) * sizeof(float);
    hipMemsetAsync(m_enc, 0, zero_bytes, stream);

    dim3 gproj(NNODES / 16, HC / 64);
    gat_proj_wmma<<<gproj, 32, 0, stream>>>(x, lin_w, xh);

    gat_att<<<(NNODES * NHEADS + 255) / 256, 256, 0, stream>>>(xh, att_src, att_dst, a_src, a_dst);

    const int eblocks = (EPRIME + 255) / 256;
    gat_edge_max<<<eblocks, 256, 0, stream>>>(ei, a_src, a_dst, m_enc);
    gat_edge_expsum<<<eblocks, 256, 0, stream>>>(ei, a_src, a_dst, m_enc, denom);
    gat_aggregate<<<EPRIME, 256, 0, stream>>>(ei, a_src, a_dst, m_enc, denom, xh, agg);

    gat_outdot<<<2048, 256, 0, stream>>>(agg, bias, out_w, ypart);
    gat_softmax<<<1, 32, 0, stream>>>(ypart, out_b, out);
}
